// TransformerBlock_9242769622343
// MI455X (gfx1250) — matrix operand order repaired and checked by reading, not compiled, not run on hardware
//
#include <hip/hip_runtime.h>
#include <hip/hip_bf16.h>

typedef _Float16 half_t;
typedef __attribute__((ext_vector_type(8)))  _Float16 v8h;
typedef __attribute__((ext_vector_type(16))) _Float16 v16h;
typedef __attribute__((ext_vector_type(8)))  float    v8f;
typedef __attribute__((ext_vector_type(4)))  int      v4i;

#define D_EMBC  1024
#define SEQC    2048
#define NTOKC   4096   // BATCH * SEQ
#define D_FFC   4096
#define D_HEADC 64

// CDNA5 async global->LDS copy (ASYNCcnt path), if this toolchain exposes it.
#if defined(__AMDGCN__) && __has_builtin(__builtin_amdgcn_global_load_async_to_lds_b128)
#define USE_ASYNC_LDS 1
#endif

__device__ __forceinline__ v8h ld8(const half_t* p) { return *(const v8h*)p; }

__device__ __forceinline__ v16h join16(v8h lo, v8h hi) {
  v16h r;
#pragma unroll
  for (int i = 0; i < 8; ++i) { r[i] = lo[i]; r[i + 8] = hi[i]; }
  return r;
}

__device__ __forceinline__ v8f wmma_f16(v16h a, v16h b, v8f c) {
  // D = A(16x32 f16) * B(32x16 f16) + C(16x16 f32)
  return __builtin_amdgcn_wmma_f32_16x16x32_f16(
      false, a, false, b, (short)0, c, false, false);
}

// A fragment, 16x32: rows m0..m0+15 of A[*][ldk], k in [kstep, kstep+32)
// Contiguous-k layout used here (not ISA table order): lane l -> row l%16; lanes<16 hold k {0..15}, lanes>=16 hold {16..31}
__device__ __forceinline__ v16h load_afrag(const half_t* A, size_t ldk,
                                           int m0, int kstep, int lane) {
  int row  = m0 + (lane & 15);
  int koff = (lane < 16) ? 0 : 16;
  const half_t* p = A + (size_t)row * ldk + kstep + koff;
  return join16(ld8(p), ld8(p + 8));
}

// ---------------------------------------------------------------- conversions
__global__ void f32_to_f16_k(const float* __restrict__ in,
                             half_t* __restrict__ out, int n) {
  int i = blockIdx.x * 256 + threadIdx.x;
  if (i < n) out[i] = (half_t)in[i];
}

// ---------------------------------------------------------------- layernorm
__global__ void __launch_bounds__(256)
layernorm_f16_k(const float* __restrict__ x, const float* __restrict__ sc,
                const float* __restrict__ sh, half_t* __restrict__ out) {
  __shared__ float red[256];
  int row = blockIdx.x;
  int t = threadIdx.x;
  const float* xr = x + (size_t)row * D_EMBC;
  float v[4];
  float s = 0.f;
#pragma unroll
  for (int i = 0; i < 4; ++i) { v[i] = xr[t + i * 256]; s += v[i]; }
  red[t] = s; __syncthreads();
  for (int o = 128; o > 0; o >>= 1) { if (t < o) red[t] += red[t + o]; __syncthreads(); }
  float mean = red[0] * (1.f / 1024.f);
  __syncthreads();
  float q = 0.f;
#pragma unroll
  for (int i = 0; i < 4; ++i) { float d = v[i] - mean; q += d * d; }
  red[t] = q; __syncthreads();
  for (int o = 128; o > 0; o >>= 1) { if (t < o) red[t] += red[t + o]; __syncthreads(); }
  float rstd = rsqrtf(red[0] * (1.f / 1024.f) + 1e-6f);
#pragma unroll
  for (int i = 0; i < 4; ++i) {
    int c = t + i * 256;
    out[(size_t)row * D_EMBC + c] = (half_t)((v[i] - mean) * rstd * sc[c] + sh[c]);
  }
}

// ---------------------------------------------------------------- GEMM (WMMA)
// C[m,n] = sum_k A[m,k] * W[n,k]  (+bias, optional gelu, +res), M%256==0, N%64==0, K%32==0
// Block = 8 waves covering 256(M) x 64(N). The 64x32 B-tile is shared by all 8 waves,
// so it is staged through LDS (double-buffered, async global->LDS when available),
// removing the 8x redundant global B reads. A stays register-direct (no cross-wave reuse).
// LDS B row stride padded to 40 halfs: 20-bank stride -> conflict-free b128 reads.
__global__ void __launch_bounds__(256)
gemm_f16_k(const half_t* __restrict__ A, const half_t* __restrict__ W,
           const float* __restrict__ bias, const float* __restrict__ res,
           float* __restrict__ outF, half_t* __restrict__ outH,
           half_t* __restrict__ outHT, int M, int N, int K, int do_gelu) {
  __shared__ __attribute__((aligned(16))) half_t bsh[2][64 * 40];
  int tid  = threadIdx.x;
  int lane = tid & 31;
  int wave = tid >> 5;
  int m0 = blockIdx.y * 256 + wave * 32;
  int n0 = blockIdx.x * 64;

  // staging role: thread -> (row 0..63, 16-byte chunk 0..3) of the B tile
  int srow = tid >> 2;
  int schk = (tid & 3) * 8;
  const half_t* gsrc_row = W + (size_t)(n0 + srow) * K + schk;
  half_t* ldst[2] = { &bsh[0][srow * 40 + schk], &bsh[1][srow * 40 + schk] };

  auto stage = [&](int buf, int k) {
    const half_t* g = gsrc_row + k;
    half_t* l = ldst[buf];
#ifdef USE_ASYNC_LDS
    __builtin_amdgcn_global_load_async_to_lds_b128(
        (__attribute__((address_space(1))) v4i*)g,
        (__attribute__((address_space(3))) v4i*)l, 0, 0);
#else
    *(v8h*)l = *(const v8h*)g;
#endif
  };

  v8f zero = {};
  v8f acc[2][4];
#pragma unroll
  for (int mi = 0; mi < 2; ++mi)
#pragma unroll
    for (int j = 0; j < 4; ++j) acc[mi][j] = zero;

  stage(0, 0);
  int buf = 0;
  for (int k = 0; k < K; k += 32, buf ^= 1) {
#ifdef USE_ASYNC_LDS
    asm volatile("s_wait_asynccnt 0x0" ::: "memory");
#endif
    __syncthreads();                       // buffer `buf` fully staged, prev reads done
    if (k + 32 < K) stage(buf ^ 1, k + 32);

    v16h a0 = load_afrag(A, (size_t)K, m0, k, lane);
    v16h a1 = load_afrag(A, (size_t)K, m0 + 16, k, lane);
    const half_t* bbase = &bsh[buf][0];
    int kbase = (lane < 16) ? 0 : 16;
#pragma unroll
    for (int j = 0; j < 4; ++j) {
      const half_t* p = bbase + (j * 16 + (lane & 15)) * 40 + kbase;
      v16h b = join16(ld8(p), ld8(p + 8));
      acc[0][j] = wmma_f16(a0, b, acc[0][j]);
      acc[1][j] = wmma_f16(a1, b, acc[1][j]);
    }
    __syncthreads();                       // all waves done reading `buf`
  }

  int lane16 = lane & 15;
  int half8 = (lane >> 4) * 8;
#pragma unroll
  for (int mi = 0; mi < 2; ++mi)
#pragma unroll
    for (int j = 0; j < 4; ++j)
#pragma unroll
      for (int r = 0; r < 8; ++r) {
        int row = m0 + mi * 16 + r + half8;
        int col = n0 + j * 16 + lane16;
        float v = acc[mi][j][r];
        if (bias) v += bias[col];
        if (do_gelu) {
          float t = tanhf(0.7978845608028654f * (v + 0.044715f * v * v * v));
          v = 0.5f * v * (1.0f + t);
        }
        size_t idx = (size_t)row * N + col;
        if (res) v += res[idx];
        if (outF) outF[idx] = v;
        if (outH) outH[idx] = (half_t)v;
        if (outHT) outHT[(size_t)col * M + row] = (half_t)v;
      }
}

// ---------------------------------------------------------------- flash attention
// Q,K: f16 [NTOK, D_EMB] (head h at col h*64); vT: f16 [D_EMB, NTOK] (transposed V).
// One wave owns a 16-query tile; streams 32 keys/step with online softmax.
__global__ void __launch_bounds__(128)
attn_flash_k(const half_t* __restrict__ Q, const half_t* __restrict__ Km,
             const half_t* __restrict__ vT, half_t* __restrict__ ctx) {
  __shared__ __attribute__((aligned(32))) half_t plds[4][16 * 32];
  int lane   = threadIdx.x & 31;
  int wave   = threadIdx.x >> 5;
  int lane16 = lane & 15;
  int hiHalf = lane >> 4;      // 0 or 1
  int half8  = hiHalf * 8;
  int koff   = hiHalf ? 16 : 0; // A-frag k offset
  int kb16   = hiHalf ? 16 : 0;// B-frag k base

  int bh = blockIdx.y;
  int b = bh >> 4, h = bh & 15;
  int q0 = blockIdx.x * 64 + wave * 16;

  // Q A-fragments (16 queries x 64 dh -> two 16x32 frags), loaded once
  const half_t* Qrow = Q + (size_t)(b * SEQC + q0 + lane16) * D_EMBC + h * D_HEADC;
  v16h qa0 = join16(ld8(Qrow + koff), ld8(Qrow + koff + 8));
  v16h qa1 = join16(ld8(Qrow + 32 + koff), ld8(Qrow + 32 + koff + 8));

  float m_run[8], l_run[8];
  v8f zero = {};
  v8f acc[4];
#pragma unroll
  for (int r = 0; r < 8; ++r) { m_run[r] = -1e30f; l_run[r] = 0.f; }
#pragma unroll
  for (int t2 = 0; t2 < 4; ++t2) acc[t2] = zero;

  half_t* my = &plds[wave][0];
  const half_t* vb_base = vT + (size_t)(h * D_HEADC + lane16) * NTOKC + b * SEQC;

  int nblk = ((q0 + 15) >> 5) + 1;   // causal: keys up to q0+15
  for (int blk = 0; blk < nblk; ++blk) {
    int kb = blk * 32;

    // ---- S = Q K^T for 32 keys (two 16-key groups) : 4 WMMAs
    v8f s0 = zero, s1 = zero;
    {
      const half_t* Kr0 = Km + (size_t)(b * SEQC + kb + lane16) * D_EMBC + h * D_HEADC + kb16;
      v16h b00 = join16(ld8(Kr0), ld8(Kr0 + 8));        // dh 0..31
      v16h b01 = join16(ld8(Kr0 + 32), ld8(Kr0 + 40));  // dh 32..63
      s0 = wmma_f16(qa0, b00, s0);
      s0 = wmma_f16(qa1, b01, s0);
      const half_t* Kr1 = Km + (size_t)(b * SEQC + kb + 16 + lane16) * D_EMBC + h * D_HEADC + kb16;
      v16h b10 = join16(ld8(Kr1), ld8(Kr1 + 8));
      v16h b11 = join16(ld8(Kr1 + 32), ld8(Kr1 + 40));
      s1 = wmma_f16(qa0, b10, s1);
      s1 = wmma_f16(qa1, b11, s1);
    }

    // ---- online softmax update (row stats live in registers; C-layout row == lane half)
    float p0[8], p1[8];
#pragma unroll
    for (int r = 0; r < 8; ++r) {
      int row = q0 + r + half8;
      float a0 = (kb + lane16      <= row) ? s0[r] * 0.125f : -1e30f;
      float a1 = (kb + 16 + lane16 <= row) ? s1[r] * 0.125f : -1e30f;
      float mx = fmaxf(a0, a1);
      mx = fmaxf(mx, __shfl_xor(mx, 1, 32));
      mx = fmaxf(mx, __shfl_xor(mx, 2, 32));
      mx = fmaxf(mx, __shfl_xor(mx, 4, 32));
      mx = fmaxf(mx, __shfl_xor(mx, 8, 32));
      float mnew  = fmaxf(m_run[r], mx);
      float alpha = __expf(m_run[r] - mnew);
      p0[r] = __expf(a0 - mnew);
      p1[r] = __expf(a1 - mnew);
      float rs = p0[r] + p1[r];
      rs += __shfl_xor(rs, 1, 32);
      rs += __shfl_xor(rs, 2, 32);
      rs += __shfl_xor(rs, 4, 32);
      rs += __shfl_xor(rs, 8, 32);
      l_run[r] = l_run[r] * alpha + rs;
      m_run[r] = mnew;
#pragma unroll
      for (int t2 = 0; t2 < 4; ++t2) acc[t2][r] = acc[t2][r] * alpha;
    }

    // ---- transpose P (C-layout f32 -> A-layout f16) through per-wave LDS
#pragma unroll
    for (int r = 0; r < 8; ++r) {
      my[(r + half8) * 32 + lane16]      = (half_t)p0[r];
      my[(r + half8) * 32 + 16 + lane16] = (half_t)p1[r];
    }
    asm volatile("s_wait_dscnt 0x0" ::: "memory");
    const half_t* pp = my + lane16 * 32 + koff;
    v16h pa = join16(*(const v8h*)pp, *(const v8h*)(pp + 8));

    // ---- O += P V : 4 WMMAs (V pre-transposed -> contiguous B-frags)
    const half_t* vp = vb_base + kb + kb16;
#pragma unroll
    for (int t2 = 0; t2 < 4; ++t2) {
      const half_t* v2 = vp + (size_t)t2 * 16 * NTOKC;
      v16h vb = join16(ld8(v2), ld8(v2 + 8));
      acc[t2] = wmma_f16(pa, vb, acc[t2]);
    }
  }

  // ---- normalize and write ctx [NTOK, D_EMB]
#pragma unroll
  for (int r = 0; r < 8; ++r) {
    float inv = 1.0f / l_run[r];
#pragma unroll
    for (int t2 = 0; t2 < 4; ++t2) {
      ctx[(size_t)(b * SEQC + q0 + r + half8) * D_EMBC + h * D_HEADC + t2 * 16 + lane16] =
          (half_t)(acc[t2][r] * inv);
    }
  }
}

// ---------------------------------------------------------------- launcher
extern "C" void kernel_launch(void* const* d_in, const int* in_sizes, int n_in,
                              void* d_out, int out_size, void* d_ws, size_t ws_size,
                              hipStream_t stream) {
  (void)in_sizes; (void)n_in; (void)out_size; (void)ws_size;
  const float* x    = (const float*)d_in[0];
  const float* wq   = (const float*)d_in[1];
  const float* wk   = (const float*)d_in[2];
  const float* wv   = (const float*)d_in[3];
  const float* wo   = (const float*)d_in[4];
  const float* bo   = (const float*)d_in[5];
  const float* w1   = (const float*)d_in[6];
  const float* b1   = (const float*)d_in[7];
  const float* w2   = (const float*)d_in[8];
  const float* b2   = (const float*)d_in[9];
  const float* ln1s = (const float*)d_in[10];
  const float* ln1b = (const float*)d_in[11];
  const float* ln2s = (const float*)d_in[12];
  const float* ln2b = (const float*)d_in[13];
  float* out = (float*)d_out;

  char* ws = (char*)d_ws;
  const size_t MB = (size_t)1 << 20;
  half_t* wq_h  = (half_t*)(ws + 0 * MB);   // 2 MB
  half_t* wk_h  = (half_t*)(ws + 2 * MB);   // 2 MB
  half_t* wv_h  = (half_t*)(ws + 4 * MB);   // 2 MB
  half_t* wo_h  = (half_t*)(ws + 6 * MB);   // 2 MB
  half_t* w1_h  = (half_t*)(ws + 8 * MB);   // 8 MB
  half_t* w2_h  = (half_t*)(ws + 16 * MB);  // 8 MB
  half_t* xn_h  = (half_t*)(ws + 24 * MB);  // 8 MB (ln1 out, later ln2 out)
  half_t* q_h   = (half_t*)(ws + 32 * MB);  // 8 MB
  half_t* k_h   = (half_t*)(ws + 40 * MB);  // 8 MB
  half_t* vT_h  = (half_t*)(ws + 48 * MB);  // 8 MB  [D_EMB, NTOK]
  half_t* ctx_h = (half_t*)(ws + 56 * MB);  // 8 MB
  float*  h_f   = (float*)(ws + 64 * MB);   // 16 MB
  half_t* y1_h  = (half_t*)(ws + 32 * MB);  // 32 MB, reuses q/k/vT/ctx region
  // total: 80 MB

  f32_to_f16_k<<<(1024 * 1024 + 255) / 256, 256, 0, stream>>>(wq, wq_h, 1024 * 1024);
  f32_to_f16_k<<<(1024 * 1024 + 255) / 256, 256, 0, stream>>>(wk, wk_h, 1024 * 1024);
  f32_to_f16_k<<<(1024 * 1024 + 255) / 256, 256, 0, stream>>>(wv, wv_h, 1024 * 1024);
  f32_to_f16_k<<<(1024 * 1024 + 255) / 256, 256, 0, stream>>>(wo, wo_h, 1024 * 1024);
  f32_to_f16_k<<<(4096 * 1024 + 255) / 256, 256, 0, stream>>>(w1, w1_h, 4096 * 1024);
  f32_to_f16_k<<<(4096 * 1024 + 255) / 256, 256, 0, stream>>>(w2, w2_h, 4096 * 1024);

  // xn1 = LN1(x)
  layernorm_f16_k<<<NTOKC, 256, 0, stream>>>(x, ln1s, ln1b, xn_h);

  dim3 gProj(1024 / 64, 4096 / 256);  // N=1024 GEMMs
  // Q, K (row layout), V (transposed layout)
  gemm_f16_k<<<gProj, 256, 0, stream>>>(xn_h, wq_h, nullptr, nullptr, nullptr, q_h, nullptr,
                                        4096, 1024, 1024, 0);
  gemm_f16_k<<<gProj, 256, 0, stream>>>(xn_h, wk_h, nullptr, nullptr, nullptr, k_h, nullptr,
                                        4096, 1024, 1024, 0);
  gemm_f16_k<<<gProj, 256, 0, stream>>>(xn_h, wv_h, nullptr, nullptr, nullptr, nullptr, vT_h,
                                        4096, 1024, 1024, 0);

  // ctx = softmax(QK^T * scale, causal) V
  attn_flash_k<<<dim3(SEQC / 64, 32), 128, 0, stream>>>(q_h, k_h, vT_h, ctx_h);

  // h = x + ctx @ wo^T + bo   (f32)
  gemm_f16_k<<<gProj, 256, 0, stream>>>(ctx_h, wo_h, bo, x, h_f, nullptr, nullptr,
                                        4096, 1024, 1024, 0);

  // xn2 = LN2(h)
  layernorm_f16_k<<<NTOKC, 256, 0, stream>>>(h_f, ln2s, ln2b, xn_h);

  // y1 = gelu(xn2 @ w1^T + b1)  (f16)
  dim3 gFF(4096 / 64, 4096 / 256);
  gemm_f16_k<<<gFF, 256, 0, stream>>>(xn_h, w1_h, b1, nullptr, nullptr, y1_h, nullptr,
                                      4096, 4096, 1024, 1);

  // out = h + y1 @ w2^T + b2   (f32)
  gemm_f16_k<<<gProj, 256, 0, stream>>>(y1_h, w2_h, b2, h_f, out, nullptr, nullptr,
                                        4096, 1024, 4096, 0);
}
